// GNNStack_88424786690722
// MI455X (gfx1250) — matrix operand order repaired and checked by reading, not compiled, not run on hardware
//
#include <hip/hip_runtime.h>

// ---------------------------------------------------------------------------
// CDNA5 (gfx1250) GNNStack forward.  wave32, WMMA bf16 16x16x32 (f32 accum),
// TDM tensor_load_to_lds staging of the A-tile in the big layer-0 GEMM.
// ---------------------------------------------------------------------------

typedef __attribute__((ext_vector_type(16))) __bf16        v16bf;
typedef __attribute__((ext_vector_type(8)))  float         v8f;
typedef __attribute__((ext_vector_type(8)))  unsigned int  v8u;
typedef __attribute__((ext_vector_type(4)))  unsigned int  u32x4;
typedef __attribute__((ext_vector_type(8)))  int           i32x8;
typedef __attribute__((ext_vector_type(4)))  int           i32x4;

#define HID 128
#define LINK 668
#define LINKP 672   // 668 padded to multiple of 32 for K-tiling
#define LROW (LINKP / 2)   // dwords per padded feature row (336)

__device__ __forceinline__ unsigned short f32_to_bf16(float f) {
  unsigned int u = __float_as_uint(f);
  unsigned int r = (u + 0x7FFFu + ((u >> 16) & 1u)) >> 16;  // round-nearest-even
  return (unsigned short)r;
}

// A-fragment (16-bit A 16x32, ISA 7.12.2): lane row = lane&15, half = lane>>4.
// VGPR0-3: K = {2v, 2v+1} + half*8 ; VGPR4-7: K = 16 + {...} + half*8.
// => dwords {o..o+3, o+8..o+11}, o = kd + half*4.
__device__ __forceinline__ v16bf ld_fragA(const unsigned int* __restrict__ row,
                                          int kd, int half) {
  v8u t;
  int o = kd + (half << 2);
  t[0] = row[o + 0];  t[1] = row[o + 1];  t[2] = row[o + 2];  t[3] = row[o + 3];
  t[4] = row[o + 8];  t[5] = row[o + 9];  t[6] = row[o + 10]; t[7] = row[o + 11];
  return __builtin_bit_cast(v16bf, t);
}

// B-fragment (16-bit B 32x16): lane col = lane&15; same K order as A-fragment:
// dwords {o..o+3, o+8..o+11}, o = kd + half*4.
__device__ __forceinline__ v16bf ld_fragB(const unsigned int* __restrict__ row,
                                          int kd, int half) {
  v8u t;
  int o = kd + (half << 2);
  #pragma unroll
  for (int i = 0; i < 8; ++i) t[i] = row[o + i + ((i >> 2) << 2)];
  return __builtin_bit_cast(v16bf, t);
}

__device__ __forceinline__ v8f wmma_bf16(v16bf a, v16bf b, v8f c) {
  return __builtin_amdgcn_wmma_f32_16x16x32_bf16(false, a, false, b,
                                                 (short)0, c, false, false);
}

// ---------------------------------------------------------------------------
// TDM: DMA a 2D bf16 tile [rows x width] (row stride = stride elems) from
// global memory into LDS at lds_off.  D# layout per CDNA5 ISA 8.3/8.4.
// ---------------------------------------------------------------------------
__device__ __forceinline__ void tdm_load_2d(unsigned lds_off, const void* gptr,
                                            int width, int rows, int tile_rows,
                                            int stride_elems) {
  unsigned long long ga = (unsigned long long)gptr;
  u32x4 g0;
  g0[0] = 1u;                                    // count=1, user-mode load
  g0[1] = lds_off;                               // lds_addr (bytes)
  g0[2] = (unsigned)(ga & 0xFFFFFFFFull);        // global_addr[31:0]
  g0[3] = (unsigned)((ga >> 32) & 0x01FFFFFFull) // global_addr[56:32]
          | 0x80000000u;                         // type=2 ("image")
  i32x8 g1;
  g1[0] = 0x10000;                               // data_size=1 (2 bytes)
  g1[1] = (int)((width & 0xFFFF) << 16);         // tensor_dim0[15:0]
  g1[2] = (int)(((width >> 16) & 0xFFFF) |
                ((rows & 0xFFFF) << 16));        // tensor_dim0 hi | tensor_dim1 lo
  g1[3] = (int)(((rows >> 16) & 0xFFFF) |
                ((width & 0xFFFF) << 16));       // tensor_dim1 hi | tile_dim0
  g1[4] = tile_rows & 0xFFFF;                    // tile_dim1 (tile_dim2 = 0)
  g1[5] = stride_elems;                          // tensor_dim0_stride[31:0]
  g1[6] = 0;
  g1[7] = 0;
  i32x4 z4 = {0, 0, 0, 0};
#if defined(__clang_major__) && (__clang_major__ >= 23)
  i32x8 z8 = {0, 0, 0, 0, 0, 0, 0, 0};
  __builtin_amdgcn_tensor_load_to_lds(g0, g1, z4, z4, z8, 0);
#else
  __builtin_amdgcn_tensor_load_to_lds(g0, g1, z4, z4, 0);
#endif
}

// ---------------------------------------------------------------------------
// Conv 3x3 VALID + bias + concat histories + ReLU -> bf16 h[N][672]
// ---------------------------------------------------------------------------
__global__ __launch_bounds__(256)
void conv_feat_kernel(const float* __restrict__ x, const float* __restrict__ hist,
                      const float* __restrict__ cw, const float* __restrict__ cb,
                      unsigned short* __restrict__ hbf, int N) {
  __shared__ float sx[8 * 11 * 11];  // 968
  __shared__ float sw[8 * 8 * 9];    // 576
  int n = blockIdx.x;
  if (n >= N) return;
  for (int i = threadIdx.x; i < 968; i += blockDim.x) sx[i] = x[(size_t)n * 968 + i];
  for (int i = threadIdx.x; i < 576; i += blockDim.x) sw[i] = cw[i];
  __syncthreads();
  for (int idx = threadIdx.x; idx < LINKP; idx += blockDim.x) {
    float v;
    if (idx < 648) {
      int o = idx / 81, rem = idx % 81, i0 = rem / 9, j0 = rem % 9;
      float s = cb[o];
      #pragma unroll
      for (int c = 0; c < 8; ++c)
        #pragma unroll
        for (int ki = 0; ki < 3; ++ki)
          #pragma unroll
          for (int kj = 0; kj < 3; ++kj)
            s += sx[c * 121 + (i0 + ki) * 11 + (j0 + kj)] *
                 sw[(o * 8 + c) * 9 + ki * 3 + kj];
      v = s;
    } else if (idx < LINK) {
      v = hist[(size_t)n * 20 + (idx - 648)];
    } else {
      v = 0.f;   // K padding
    }
    v = fmaxf(v, 0.f);
    hbf[(size_t)n * LINKP + idx] = f32_to_bf16(v);
  }
}

// f32 weight [K][Nn] -> bf16 transposed [Nn][Kpad] (zero-padded K)
__global__ __launch_bounds__(256)
void cvt_wT_kernel(const float* __restrict__ W, unsigned short* __restrict__ O,
                   int K, int Kpad, int Nn) {
  int i = blockIdx.x * blockDim.x + threadIdx.x;
  if (i >= Nn * Kpad) return;
  int n = i / Kpad, k = i % Kpad;
  float v = (k < K) ? W[(size_t)k * Nn + n] : 0.f;
  O[i] = f32_to_bf16(v);
}

__global__ __launch_bounds__(256)
void zero_f32_kernel(float* __restrict__ p, long long count) {
  long long i = (long long)blockIdx.x * blockDim.x + threadIdx.x;
  if (i < count) p[i] = 0.f;
}

__global__ __launch_bounds__(256)
void cvt_f32_bf16_kernel(const float* __restrict__ in,
                         unsigned short* __restrict__ out, long long count) {
  long long i = (long long)blockIdx.x * blockDim.x + threadIdx.x;
  if (i < count) out[i] = f32_to_bf16(in[i]);
}

// agg_bf16 = bf16(agg / max(deg,1))
__global__ __launch_bounds__(256)
void cvt_agg_kernel(const float* __restrict__ agg, const float* __restrict__ deg,
                    unsigned short* __restrict__ out, long long count) {
  long long i = (long long)blockIdx.x * blockDim.x + threadIdx.x;
  if (i >= count) return;
  float d = fmaxf(deg[i >> 7], 1.f);
  out[i] = f32_to_bf16(agg[i] / d);
}

// ---------------------------------------------------------------------------
// Edge scatter: dst_buf[dst] += src_feat[src] (128 feats, f32 atomics in L2).
// ---------------------------------------------------------------------------
__global__ __launch_bounds__(256)
void scatter_add_kernel(const int* __restrict__ ei, const float* __restrict__ src_feat,
                        float* __restrict__ dst_buf, float* __restrict__ deg,
                        int E, int mask_self) {
  long long t = (long long)blockIdx.x * blockDim.x + threadIdx.x;
  int e = (int)(t >> 5);
  if (e >= E) return;
  int lane = (int)(t & 31);
  int s = ei[e], d = ei[(size_t)E + e];
  if (deg != nullptr && lane == 0) atomicAdd(&deg[d], 1.0f);
  if (mask_self && s == d) return;
  const float4 v = *(const float4*)(src_feat + (size_t)s * HID + lane * 4);
  float* dp = dst_buf + (size_t)d * HID + lane * 4;
  atomicAdd(dp + 0, v.x); atomicAdd(dp + 1, v.y);
  atomicAdd(dp + 2, v.z); atomicAdd(dp + 3, v.w);
}

// ---------------------------------------------------------------------------
// Layer-0 GEMM: selfx = h@Wself + bs ; xn = h@W + bx.  K=672.
// Wave 0 TDM-stages the 16x672 bf16 A-tile into LDS once per block; all 8
// waves (one 16x16 output tile each) read A-fragments from LDS.
// ---------------------------------------------------------------------------
__global__ __launch_bounds__(256)
void gemm_lin_kernel(const unsigned short* __restrict__ hbf,
                     const unsigned short* __restrict__ wsT,
                     const unsigned short* __restrict__ wT,
                     const float* __restrict__ bs, const float* __restrict__ bx,
                     float* __restrict__ selfx, float* __restrict__ xn, int N) {
  __shared__ unsigned int sA[16 * LROW];   // 21504 B
  int tm   = blockIdx.x;
  int wv   = threadIdx.x >> 5;
  int lane = threadIdx.x & 31;
  int half = lane >> 4, l16 = lane & 15;

  if (wv == 0) {
    int rows_left = N - tm * 16;           // tensor rows from tile start (OOB->0)
    tdm_load_2d((unsigned)(size_t)(void*)&sA[0],
                hbf + (size_t)tm * 16 * LINKP,
                LINKP, rows_left, 16, LINKP);
    __builtin_amdgcn_s_wait_tensorcnt(0);
  }
  __syncthreads();

  const unsigned int* arow = &sA[l16 * LROW];
  const unsigned int* b0r  = (const unsigned int*)(wsT + (size_t)(wv * 16 + l16) * LINKP);
  const unsigned int* b1r  = (const unsigned int*)(wT  + (size_t)(wv * 16 + l16) * LINKP);
  v8f acc0 = {}; v8f acc1 = {};
  for (int kk = 0; kk < LINKP / 32; ++kk) {
    v16bf a  = ld_fragA(arow, kk * 16, half);
    v16bf w0 = ld_fragB(b0r,  kk * 16, half);
    v16bf w1 = ld_fragB(b1r,  kk * 16, half);
    acc0 = wmma_bf16(a, w0, acc0);
    acc1 = wmma_bf16(a, w1, acc1);
  }
  int col = wv * 16 + l16;
  float bb0 = bs[col], bb1 = bx[col];
  int rb = tm * 16 + half * 8;
  #pragma unroll
  for (int r = 0; r < 8; ++r) {
    int row = rb + r;
    if (row < N) {
      selfx[(size_t)row * HID + col] = acc0[r] + bb0;
      xn   [(size_t)row * HID + col] = acc1[r] + bb1;
    }
  }
}

// SAGE GEMM: out = agg@wl + hidden@wr + bl.  K=128, chained accumulator.
__global__ __launch_bounds__(256)
void gemm_sage_kernel(const unsigned short* __restrict__ abf,
                      const unsigned short* __restrict__ hbf,
                      const unsigned short* __restrict__ wlT,
                      const unsigned short* __restrict__ wrT,
                      const float* __restrict__ bl,
                      float* __restrict__ out, int N) {
  int tm = blockIdx.x, wv = threadIdx.x >> 5, lane = threadIdx.x & 31;
  int half = lane >> 4, l16 = lane & 15;
  int ar = tm * 16 + l16;  if (ar >= N) ar = N - 1;
  const unsigned int* a0r = (const unsigned int*)(abf + (size_t)ar * HID);
  const unsigned int* a1r = (const unsigned int*)(hbf + (size_t)ar * HID);
  const unsigned int* b0r = (const unsigned int*)(wlT + (size_t)(wv * 16 + l16) * HID);
  const unsigned int* b1r = (const unsigned int*)(wrT + (size_t)(wv * 16 + l16) * HID);
  v8f acc = {};
  #pragma unroll
  for (int kk = 0; kk < HID / 32; ++kk) {
    acc = wmma_bf16(ld_fragA(a0r, kk * 16, half), ld_fragB(b0r, kk * 16, half), acc);
    acc = wmma_bf16(ld_fragA(a1r, kk * 16, half), ld_fragB(b1r, kk * 16, half), acc);
  }
  int col = wv * 16 + l16;
  float bb = bl[col];
  int rb = tm * 16 + half * 8;
  #pragma unroll
  for (int r = 0; r < 8; ++r) {
    int row = rb + r;
    if (row < N) out[(size_t)row * HID + col] = acc[r] + bb;
  }
}

// Single GEMM: out = h@W + b.  K=128.
__global__ __launch_bounds__(256)
void gemm_one_kernel(const unsigned short* __restrict__ hbf,
                     const unsigned short* __restrict__ wT,
                     const float* __restrict__ bias,
                     float* __restrict__ out, int N) {
  int tm = blockIdx.x, wv = threadIdx.x >> 5, lane = threadIdx.x & 31;
  int half = lane >> 4, l16 = lane & 15;
  int ar = tm * 16 + l16;  if (ar >= N) ar = N - 1;
  const unsigned int* arow = (const unsigned int*)(hbf + (size_t)ar * HID);
  const unsigned int* brow = (const unsigned int*)(wT + (size_t)(wv * 16 + l16) * HID);
  v8f acc = {};
  #pragma unroll
  for (int kk = 0; kk < HID / 32; ++kk)
    acc = wmma_bf16(ld_fragA(arow, kk * 16, half), ld_fragB(brow, kk * 16, half), acc);
  int col = wv * 16 + l16;
  float bb = bias[col];
  int rb = tm * 16 + half * 8;
  #pragma unroll
  for (int r = 0; r < 8; ++r) {
    int row = rb + r;
    if (row < N) out[(size_t)row * HID + col] = acc[r] + bb;
  }
}

// ---------------------------------------------------------------------------
// hidden = LN(relu(out)) (do_ln) or relu(out); optionally emb = out (raw).
// ---------------------------------------------------------------------------
__global__ __launch_bounds__(128)
void relu_ln_kernel(const float* __restrict__ out, const float* __restrict__ g,
                    const float* __restrict__ b, float* __restrict__ hidden,
                    float* __restrict__ emb, int do_ln, int N) {
  __shared__ float s[HID];
  int n = blockIdx.x;
  if (n >= N) return;
  int t = threadIdx.x;
  float o = out[(size_t)n * HID + t];
  if (emb != nullptr) emb[(size_t)n * HID + t] = o;
  float v = fmaxf(o, 0.f);
  if (do_ln) {
    s[t] = v;  __syncthreads();
    for (int st = 64; st > 0; st >>= 1) { if (t < st) s[t] += s[t + st]; __syncthreads(); }
    float m = s[0] * (1.f / HID);  __syncthreads();
    float d = v - m;
    s[t] = d * d;  __syncthreads();
    for (int st = 64; st > 0; st >>= 1) { if (t < st) s[t] += s[t + st]; __syncthreads(); }
    float var = s[0] * (1.f / HID);
    v = d * rsqrtf(var + 1e-5f) * g[t] + b[t];
  }
  hidden[(size_t)n * HID + t] = v;
}

// logits = t@mp_w2 + b2 ; d_out = log_softmax(logits).
__global__ __launch_bounds__(256)
void post_logits_kernel(const float* __restrict__ tbuf, const float* __restrict__ w2,
                        const float* __restrict__ b2, float* __restrict__ outls, int N) {
  int n = blockIdx.x * blockDim.x + threadIdx.x;
  if (n >= N) return;
  float lg[5];
  #pragma unroll
  for (int o = 0; o < 5; ++o) lg[o] = b2[o];
  for (int k = 0; k < HID; ++k) {
    float tv = tbuf[(size_t)n * HID + k];
    #pragma unroll
    for (int o = 0; o < 5; ++o) lg[o] += tv * w2[k * 5 + o];
  }
  float mx = lg[0];
  #pragma unroll
  for (int o = 1; o < 5; ++o) mx = fmaxf(mx, lg[o]);
  float se = 0.f;
  #pragma unroll
  for (int o = 0; o < 5; ++o) se += __expf(lg[o] - mx);
  float lse = logf(se) + mx;
  #pragma unroll
  for (int o = 0; o < 5; ++o) outls[(size_t)n * 5 + o] = lg[o] - lse;
}

// ---------------------------------------------------------------------------
extern "C" void kernel_launch(void* const* d_in, const int* in_sizes, int n_in,
                              void* d_out, int out_size, void* d_ws, size_t ws_size,
                              hipStream_t stream) {
  const float* x          = (const float*)d_in[0];
  const float* hist       = (const float*)d_in[1];
  const float* conv_w     = (const float*)d_in[2];
  const float* conv_b     = (const float*)d_in[3];
  const float* lin_self_w = (const float*)d_in[4];
  const float* lin_self_b = (const float*)d_in[5];
  const float* lin_w      = (const float*)d_in[6];
  const float* lin_b      = (const float*)d_in[7];
  const float* sage_wl[3] = {(const float*)d_in[8],  (const float*)d_in[11], (const float*)d_in[14]};
  const float* sage_bl[3] = {(const float*)d_in[9],  (const float*)d_in[12], (const float*)d_in[15]};
  const float* sage_wr[3] = {(const float*)d_in[10], (const float*)d_in[13], (const float*)d_in[16]};
  const float* ln_g[2]    = {(const float*)d_in[17], (const float*)d_in[19]};
  const float* ln_b[2]    = {(const float*)d_in[18], (const float*)d_in[20]};
  const float* mp_w1      = (const float*)d_in[21];
  const float* mp_b1      = (const float*)d_in[22];
  const float* mp_w2      = (const float*)d_in[23];
  const float* mp_b2      = (const float*)d_in[24];
  const int*   ei         = (const int*)d_in[25];

  const int N = in_sizes[0] / (8 * 11 * 11);
  const int E = in_sizes[25] / 2;
  const long long NH = (long long)N * HID;

  // -------- workspace cursor allocation --------
  char* cur = (char*)d_ws;
  auto alloc = [&](size_t bytes) {
    char* p = cur;
    cur += (bytes + 255) & ~(size_t)255;
    return (void*)p;
  };
  unsigned short* hbf   = (unsigned short*)alloc((size_t)N * LINKP * 2);
  float*          hidden= (float*)alloc((size_t)N * HID * 4);   // selfx -> hidden
  float*          xn    = (float*)alloc((size_t)N * HID * 4);   // also t in post_mp
  float*          agg   = (float*)alloc((size_t)N * HID * 4);
  float*          outb  = (float*)alloc((size_t)N * HID * 4);
  unsigned short* hbf2  = (unsigned short*)alloc((size_t)N * HID * 2);
  unsigned short* abf   = (unsigned short*)alloc((size_t)N * HID * 2);
  float*          deg   = (float*)alloc((size_t)N * 4);
  unsigned short* wsT   = (unsigned short*)alloc((size_t)HID * LINKP * 2);
  unsigned short* wT    = (unsigned short*)alloc((size_t)HID * LINKP * 2);
  unsigned short* wlT[3], *wrT[3];
  for (int i = 0; i < 3; ++i) {
    wlT[i] = (unsigned short*)alloc((size_t)HID * HID * 2);
    wrT[i] = (unsigned short*)alloc((size_t)HID * HID * 2);
  }
  unsigned short* w1T = (unsigned short*)alloc((size_t)HID * HID * 2);

  float* emb_out = (float*)d_out;             // [N,128]
  float* ls_out  = (float*)d_out + NH;        // [N,5]

  const int TB = 256;
  auto grid1 = [&](long long cnt) { return (unsigned)((cnt + TB - 1) / TB); };

  // -------- weight conversion (f32 -> bf16, transposed, K-padded) --------
  cvt_wT_kernel<<<grid1((long long)HID * LINKP), TB, 0, stream>>>(lin_self_w, wsT, LINK, LINKP, HID);
  cvt_wT_kernel<<<grid1((long long)HID * LINKP), TB, 0, stream>>>(lin_w,      wT,  LINK, LINKP, HID);
  for (int i = 0; i < 3; ++i) {
    cvt_wT_kernel<<<grid1((long long)HID * HID), TB, 0, stream>>>(sage_wl[i], wlT[i], HID, HID, HID);
    cvt_wT_kernel<<<grid1((long long)HID * HID), TB, 0, stream>>>(sage_wr[i], wrT[i], HID, HID, HID);
  }
  cvt_wT_kernel<<<grid1((long long)HID * HID), TB, 0, stream>>>(mp_w1, w1T, HID, HID, HID);

  // -------- conv + concat + relu -> h bf16 --------
  conv_feat_kernel<<<N, TB, 0, stream>>>(x, hist, conv_w, conv_b, hbf, N);

  // -------- layer-0 dense: selfx(->hidden) and xn --------
  int tilesM = (N + 15) / 16;
  gemm_lin_kernel<<<tilesM, TB, 0, stream>>>(hbf, wsT, wT, lin_self_b, lin_b,
                                             hidden, xn, N);

  // -------- degree + masked scatter: hidden += segsum(xn[src]) --------
  zero_f32_kernel<<<grid1(N), TB, 0, stream>>>(deg, N);
  scatter_add_kernel<<<grid1((long long)E * 32), TB, 0, stream>>>(ei, xn, hidden, deg, E, 1);

  // -------- 3 SAGE layers --------
  for (int i = 0; i < 3; ++i) {
    cvt_f32_bf16_kernel<<<grid1(NH), TB, 0, stream>>>(hidden, hbf2, NH);
    zero_f32_kernel<<<grid1(NH), TB, 0, stream>>>(agg, NH);
    scatter_add_kernel<<<grid1((long long)E * 32), TB, 0, stream>>>(ei, hidden, agg, nullptr, E, 0);
    cvt_agg_kernel<<<grid1(NH), TB, 0, stream>>>(agg, deg, abf, NH);
    gemm_sage_kernel<<<tilesM, TB, 0, stream>>>(abf, hbf2, wlT[i], wrT[i], sage_bl[i], outb, N);
    if (i < 2) {
      relu_ln_kernel<<<N, HID, 0, stream>>>(outb, ln_g[i], ln_b[i], hidden, nullptr, 1, N);
    } else {
      relu_ln_kernel<<<N, HID, 0, stream>>>(outb, nullptr, nullptr, hidden, emb_out, 0, N);
    }
  }

  // -------- post_mp --------
  cvt_f32_bf16_kernel<<<grid1(NH), TB, 0, stream>>>(hidden, hbf2, NH);
  gemm_one_kernel<<<tilesM, TB, 0, stream>>>(hbf2, w1T, mp_b1, xn, N);  // t -> xn
  post_logits_kernel<<<grid1(N), TB, 0, stream>>>(xn, mp_w2, mp_b2, ls_out, N);
}
